// LoRAMLP4RANF_84988812853866
// MI455X (gfx1250) — compile-verified, matrix operand order checked by reading, not run on hardware
//
#include <hip/hip_runtime.h>
#include <hip/hip_bf16.h>
#include <math.h>

// ---- types ----------------------------------------------------------------
typedef __bf16 v16bf  __attribute__((ext_vector_type(16)));
typedef __bf16 v8bf   __attribute__((ext_vector_type(8)));
typedef __bf16 bf16x4 __attribute__((ext_vector_type(4)));
typedef float  v8f    __attribute__((ext_vector_type(8)));

// ---- problem constants ----------------------------------------------------
constexpr int kB   = 32;
constexpr int kF   = 512;
constexpr int kIN  = 1024;
constexpr int kOUT = 1024;
constexpr int kR   = 16;

// ---- GEMM tiling ----------------------------------------------------------
constexpr int BM   = 128;   // F tile
constexpr int BN   = 128;   // OUT tile
constexpr int BK   = 32;    // bf16 WMMA K
constexpr int LSTR = 40;    // LDS row stride in bf16 elems (32 data + 8 pad)
constexpr int NK   = kIN / BK + 1;  // 32 main K-tiles + 1 LoRA tail tile

// ===========================================================================
// Pass 1: z1[b,f,r] = (1/(IN*R)) * sum_i x[b,f,i] * v[b,i,r]
// ===========================================================================
__global__ __launch_bounds__(256) void z1_kernel(const float* __restrict__ x,
                                                 const float* __restrict__ v,
                                                 float* __restrict__ z1) {
  const int b     = blockIdx.y;
  const int fBase = blockIdx.x * 64;
  const int tid   = threadIdx.x;
  const int r     = tid & 15;
  const int fl    = tid >> 4;
  const float scale = 1.0f / (float)(kIN * kR);
  const float* vb = v + (size_t)b * kIN * kR;

  for (int fo = 0; fo < 64; fo += 16) {
    const int f = fBase + fo + fl;
    const float4* xrow = (const float4*)(x + ((size_t)b * kF + f) * kIN);
    float s = 0.f;
#pragma unroll 4
    for (int i4 = 0; i4 < kIN / 4; ++i4) {
      const float4 xv = xrow[i4];
      const float* vp = vb + (size_t)i4 * 4 * kR + r;
      s += xv.x * vp[0] + xv.y * vp[kR] + xv.z * vp[2 * kR] + xv.w * vp[3 * kR];
    }
    z1[((size_t)b * kF + f) * kR + r] = s * scale;
  }
}

// ===========================================================================
// Pass 2: out = gelu( x @ W^T  +  z1 @ u^T  + fc_bias + b )
// Double-buffered bf16 WMMA GEMM over K = IN + 32 (tail tile = LoRA rank-16).
// 256 threads (8 waves), 128x128 tile, each wave 64x32 via 4x2 WMMA tiles.
// Pipeline: global loads for tile k+1 issued before the WMMAs of tile k;
// cvt+ds_store to the alternate buffer after; one barrier per iteration.
// ===========================================================================
__global__ __launch_bounds__(256) void lora_gemm_kernel(
    const float* __restrict__ x,    const float* __restrict__ u,
    const float* __restrict__ z1,   const float* __restrict__ bvec,
    const float* __restrict__ W,    const float* __restrict__ fcb,
    float* __restrict__ out) {
  __shared__ __align__(16) __bf16 As[2][BM * LSTR];
  __shared__ __align__(16) __bf16 Bs[2][BN * LSTR];

  const int b     = blockIdx.z;
  const int fBase = blockIdx.x * BM;
  const int oBase = blockIdx.y * BN;
  const int tid   = threadIdx.x;
  const int lane  = tid & 31;
  const int wave  = tid >> 5;
  const int wm    = (wave >> 2) * 64;   // wave M offset in tile (2 waves)
  const int wn    = (wave & 3)  * 32;   // wave N offset in tile (4 waves)
  const int lN    = lane & 15;
  const int kh    = lane >> 4;

  const v8f vzero = {0.f, 0.f, 0.f, 0.f, 0.f, 0.f, 0.f, 0.f};
  v8f acc[4][2];
#pragma unroll
  for (int sm = 0; sm < 4; ++sm)
#pragma unroll
    for (int sn = 0; sn < 2; ++sn) acc[sm][sn] = vzero;

  // staging map: each thread owns (row = tid>>3 in 4 sweeps of 32, col = 4 elems)
  const int ldRow = tid >> 3;       // 0..31
  const int ldCol = (tid & 7) * 4;  // 0,4,...,28

  const float* xBase = x  + ((size_t)b * kF   + fBase) * kIN;
  const float* wBase = W  +  (size_t)oBase * kIN;
  const float* zBase = z1 + ((size_t)b * kF   + fBase) * kR;
  const float* uBase = u  + ((size_t)b * kOUT + oBase) * kR;

  // in-flight staging registers (distinct per sweep -> all 8 loads overlap)
  float4 ra[4], rb[4];

  auto loadTile = [&](int kti) {
    if (kti < kIN / BK) {
      const int kt = kti * BK + ldCol;
#pragma unroll
      for (int i = 0; i < 4; ++i) {
        const int row = ldRow + i * 32;
        ra[i] = *(const float4*)(xBase + (size_t)row * kIN + kt);
        rb[i] = *(const float4*)(wBase + (size_t)row * kIN + kt);
      }
    } else {  // LoRA tail: cols 0..15 = z1 / u, cols 16..31 = zero pad
#pragma unroll
      for (int i = 0; i < 4; ++i) {
        const int row = ldRow + i * 32;
        if (ldCol < kR) {
          ra[i] = *(const float4*)(zBase + (size_t)row * kR + ldCol);
          rb[i] = *(const float4*)(uBase + (size_t)row * kR + ldCol);
        } else {
          ra[i] = make_float4(0.f, 0.f, 0.f, 0.f);
          rb[i] = make_float4(0.f, 0.f, 0.f, 0.f);
        }
      }
    }
  };

  auto storeTile = [&](int buf) {
#pragma unroll
    for (int i = 0; i < 4; ++i) {
      const int row = ldRow + i * 32;
      const bf16x4 av = {(__bf16)ra[i].x, (__bf16)ra[i].y, (__bf16)ra[i].z, (__bf16)ra[i].w};
      const bf16x4 wv = {(__bf16)rb[i].x, (__bf16)rb[i].y, (__bf16)rb[i].z, (__bf16)rb[i].w};
      *(bf16x4*)(&As[buf][row * LSTR + ldCol]) = av;
      *(bf16x4*)(&Bs[buf][row * LSTR + ldCol]) = wv;
    }
  };

  // prologue: stage tile 0
  loadTile(0);
  storeTile(0);
  __syncthreads();

#pragma unroll 2
  for (int kti = 0; kti < NK; ++kti) {
    const int cur = kti & 1;

    // issue next tile's global loads now; wait happens in storeTile below
    if (kti + 1 < NK) loadTile(kti + 1);

    // fragments from current LDS buffer
    // (ISA 16-bit A/B layout: lane = row%16, K half selected by lane/16)
    v16bf afrag[4], bfrag[2];
#pragma unroll
    for (int sm = 0; sm < 4; ++sm) {
      const __bf16* p = &As[cur][(wm + sm * 16 + lN) * LSTR + kh * 8];
      ((v8bf*)&afrag[sm])[0] = *(const v8bf*)p;          // K = kh*8 .. +7
      ((v8bf*)&afrag[sm])[1] = *(const v8bf*)(p + 16);   // K = 16+kh*8 .. +7
    }
#pragma unroll
    for (int sn = 0; sn < 2; ++sn) {
      const __bf16* p = &Bs[cur][(wn + sn * 16 + lN) * LSTR + kh * 8];
      ((v8bf*)&bfrag[sn])[0] = *(const v8bf*)p;
      ((v8bf*)&bfrag[sn])[1] = *(const v8bf*)(p + 16);
    }

#pragma unroll
    for (int sm = 0; sm < 4; ++sm)
#pragma unroll
      for (int sn = 0; sn < 2; ++sn)
        acc[sm][sn] = __builtin_amdgcn_wmma_f32_16x16x32_bf16(
            false, afrag[sm], false, bfrag[sn], (short)0, acc[sm][sn], false, false);

    // convert + stage next tile into the alternate buffer
    if (kti + 1 < NK) storeTile(cur ^ 1);
    __syncthreads();
  }

  // ---- epilogue: + fc_bias + b, exact GELU, store
  // C layout: VGPR r -> M = r + 8*(lane/16), N = lane%16
#pragma unroll
  for (int sn = 0; sn < 2; ++sn) {
    const int o = oBase + wn + sn * 16 + lN;
    const float base = fcb[o] + bvec[(size_t)b * kOUT + o];
#pragma unroll
    for (int sm = 0; sm < 4; ++sm) {
#pragma unroll
      for (int r = 0; r < 8; ++r) {
        const int f = fBase + wm + sm * 16 + kh * 8 + r;
        const float y = acc[sm][sn][r] + base;
        const float g = 0.5f * y * (1.0f + erff(y * 0.70710678118654752f));
        out[((size_t)b * kF + f) * kOUT + o] = g;
      }
    }
  }
}

// ===========================================================================
extern "C" void kernel_launch(void* const* d_in, const int* in_sizes, int n_in,
                              void* d_out, int out_size, void* d_ws, size_t ws_size,
                              hipStream_t stream) {
  (void)in_sizes; (void)n_in; (void)out_size; (void)ws_size;
  const float* x    = (const float*)d_in[0];  // [B,F,IN]
  const float* u    = (const float*)d_in[1];  // [B,1,OUT,R]
  const float* v    = (const float*)d_in[2];  // [B,1,IN,R]
  const float* bvec = (const float*)d_in[3];  // [B,1,OUT]
  const float* W    = (const float*)d_in[4];  // [OUT,IN]
  const float* fcb  = (const float*)d_in[5];  // [OUT]
  float*       out  = (float*)d_out;          // [B,F,OUT]
  float*       z1   = (float*)d_ws;           // [B,F,R] scratch (1 MB)

  dim3 g1(kF / 64, kB);
  z1_kernel<<<g1, 256, 0, stream>>>(x, v, z1);

  dim3 g2(kF / BM, kOUT / BN, kB);
  lora_gemm_kernel<<<g2, 256, 0, stream>>>(x, u, z1, bvec, W, fcb, out);
}
